// RQ_VAE_76776835383754
// MI455X (gfx1250) — matrix operand order repaired and checked by reading, not compiled, not run on hardware
//
#include <hip/hip_runtime.h>

// ---------------------------------------------------------------------------
// Types for WMMA (gfx1250, wave32)
// ---------------------------------------------------------------------------
typedef __bf16 bf16;
typedef __bf16 v16bf __attribute__((ext_vector_type(16)));
typedef __bf16 v8bf  __attribute__((ext_vector_type(8)));
typedef float  v8f   __attribute__((ext_vector_type(8)));

static __device__ __forceinline__ bf16 f2bf(float f) {
    unsigned u = __builtin_bit_cast(unsigned, f);
    unsigned r = (u + 0x7FFFu + ((u >> 16) & 1u)) >> 16;
    unsigned short s = (unsigned short)r;
    return __builtin_bit_cast(bf16, s);
}

// CDNA5 async global->LDS copy (ASYNCcnt-tracked), bypasses VGPRs.
// dst LDS address = low 32 bits of generic shared pointer (aperture rule).
static __device__ __forceinline__ void async_ld_b128(void* lds_ptr,
                                                     const void* gptr) {
    unsigned lds = (unsigned)(unsigned long long)lds_ptr;
    unsigned long long ga = (unsigned long long)gptr;
    asm volatile("global_load_async_to_lds_b128 %0, %1, off"
                 :: "v"(lds), "v"(ga)
                 : "memory");
}
static __device__ __forceinline__ void wait_async0() {
    asm volatile("s_wait_asynccnt 0x0" ::: "memory");
}

// ---------------------------------------------------------------------------
// Problem constants
// ---------------------------------------------------------------------------
#define BATCH    8192
#define IN_DIM   2048
#define EDIM     256
#define H1D      1024
#define H2D      512
#define KCB      1024
#define NLAYERS  3
#define TEMP_INV 0.8f     // 1/1.25
#define COMMIT_W 0.25f

// ---------------------------------------------------------------------------
// f32 -> bf16 flat convert
// ---------------------------------------------------------------------------
__global__ __launch_bounds__(256)
void k_f2bf(const float* __restrict__ in, bf16* __restrict__ out, long n) {
    long i = (long)blockIdx.x * 256 + threadIdx.x;
    long stride = (long)gridDim.x * 256;
    for (; i < n; i += stride) out[i] = f2bf(in[i]);
}

// transpose-convert: in[K][N] f32 -> out[N][K] bf16
__global__ __launch_bounds__(256)
void k_transpose_bf(const float* __restrict__ in, bf16* __restrict__ out,
                    int K, int N) {
    long total = (long)K * N;
    long i = (long)blockIdx.x * 256 + threadIdx.x;
    long stride = (long)gridDim.x * 256;
    for (; i < total; i += stride) {
        int k = (int)(i / N), n = (int)(i % N);
        out[(long)n * K + k] = f2bf(in[i]);
    }
}

// per-row squared norm (D == 256), one block per row
__global__ __launch_bounds__(256)
void k_rownorm2(const float* __restrict__ in, float* __restrict__ out, int D) {
    __shared__ float sbuf[256];
    int row = blockIdx.x;
    float v = in[(long)row * D + threadIdx.x];
    sbuf[threadIdx.x] = v * v; __syncthreads();
    for (int o = 128; o > 0; o >>= 1) {
        if (threadIdx.x < o) sbuf[threadIdx.x] += sbuf[threadIdx.x + o];
        __syncthreads();
    }
    if (threadIdx.x == 0) out[row] = sbuf[0];
}

// ---------------------------------------------------------------------------
// bf16 WMMA GEMM:  C[M,N] = A[M,K] @ B[K,N]   (Bt is B transposed: [N][K])
// block tile 128x64, BK=32, 8 waves, each wave 32x32 (2x2 WMMA tiles).
// Double-buffered LDS fed by CDNA5 async global->LDS loads (ASYNCcnt),
// one workgroup barrier per K-step.  act: 0 = none, 1 = SiLU
// ---------------------------------------------------------------------------
#define BM 128
#define BN 64
#define BKK 32

__global__ __launch_bounds__(256)
void k_gemm_bf16_wmma(const bf16* __restrict__ A, const bf16* __restrict__ Bt,
                      float* __restrict__ C, int M, int N, int K, int act) {
    __shared__ __align__(64) bf16 Asmem[2][BM][BKK];   // 2 x 8 KB
    __shared__ __align__(64) bf16 Bsmem[2][BN][BKK];   // 2 x 4 KB

    const int tid  = threadIdx.x;
    const int lane = tid & 31;
    const int wave = tid >> 5;        // 0..7
    const int wm   = wave & 3;        // 4 waves along M
    const int wn   = wave >> 2;       // 2 waves along N
    const int m0   = blockIdx.y * BM;
    const int n0   = blockIdx.x * BN;
    const int sel  = lane >> 4;       // 0 / 1 (lane half)
    const int l16  = lane & 15;

    // per-thread staging assignment
    const int ar = tid >> 1, ac = (tid & 1) * 16;   // A: 32 B / thread
    const int br = tid >> 2, bc = (tid & 3) * 8;    // B: 16 B / thread
    const bf16* Ag = A  + (long)(m0 + ar) * K + ac;
    const bf16* Bg = Bt + (long)(n0 + br) * K + bc;

    v8f acc[2][2] = {};

    // prologue: async-stage first K tile into buffer 0
    async_ld_b128(&Asmem[0][ar][ac],     Ag);
    async_ld_b128(&Asmem[0][ar][ac + 8], Ag + 8);
    async_ld_b128(&Bsmem[0][br][bc],     Bg);

    int buf = 0;
    for (int k0 = 0; k0 < K; k0 += BKK) {
        wait_async0();        // my wave's async tile landed in LDS
        __syncthreads();      // everyone's tile landed

        if (k0 + BKK < K) {   // overlap: issue next tile into other buffer
            async_ld_b128(&Asmem[buf ^ 1][ar][ac],     Ag + k0 + BKK);
            async_ld_b128(&Asmem[buf ^ 1][ar][ac + 8], Ag + k0 + BKK + 8);
            async_ld_b128(&Bsmem[buf ^ 1][br][bc],     Bg + k0 + BKK);
        }

        // A frags per ISA 16-bit 16x32 layout:
        //   lane<16: K = {0..7} u {16..23};  lane>=16: K = {8..15} u {24..31}
        v16bf afrag[2], bfrag[2];
        for (int r = 0; r < 2; ++r) {
            int m   = wm * 32 + r * 16 + l16;
            int klo = sel * 8;
            v8bf lo = *(const v8bf*)(&Asmem[buf][m][klo]);
            v8bf hi = *(const v8bf*)(&Asmem[buf][m][klo + 16]);
            for (int i = 0; i < 8; ++i) { afrag[r][i] = lo[i]; afrag[r][i + 8] = hi[i]; }
        }
        // B frags: lane<16 -> K={0..7} u {16..23}, lane>=16 -> K={8..15} u {24..31}, col = lane%16
        for (int c = 0; c < 2; ++c) {
            int n = wn * 32 + c * 16 + l16;
            v8bf lo = *(const v8bf*)(&Bsmem[buf][n][sel * 8]);
            v8bf hi = *(const v8bf*)(&Bsmem[buf][n][sel * 8 + 16]);
            for (int i = 0; i < 8; ++i) { bfrag[c][i] = lo[i]; bfrag[c][i + 8] = hi[i]; }
        }
        for (int r = 0; r < 2; ++r)
            for (int c = 0; c < 2; ++c)
                acc[r][c] = __builtin_amdgcn_wmma_f32_16x16x32_bf16(
                    false, afrag[r], false, bfrag[c],
                    (short)0, acc[r][c], false, false);
        buf ^= 1;
        // no trailing barrier: buffer buf^1 (the one async is writing) was
        // last read one K-step ago, strictly before the issuing barrier.
    }

    // epilogue: C/D layout -> VGPR i : row = i + 8*sel, col = l16
    for (int r = 0; r < 2; ++r)
        for (int c = 0; c < 2; ++c) {
            int n = n0 + wn * 32 + c * 16 + l16;
            for (int i = 0; i < 8; ++i) {
                int m = m0 + wm * 32 + r * 16 + sel * 8 + i;
                float v = acc[r][c][i];
                if (act == 1) v = v / (1.0f + __expf(-v));   // SiLU
                C[(long)m * N + n] = v;
            }
        }
}

// ---------------------------------------------------------------------------
// LayerNorm (biased var) f32 -> bf16, one block per row
// ---------------------------------------------------------------------------
__global__ __launch_bounds__(256)
void k_layernorm_bf(const float* __restrict__ in, const float* __restrict__ g,
                    const float* __restrict__ b, bf16* __restrict__ out, int H) {
    __shared__ float s1[256], s2[256];
    int row = blockIdx.x, tid = threadIdx.x;
    const float* x = in + (long)row * H;
    float a = 0.f, q = 0.f;
    for (int i = tid; i < H; i += 256) { float v = x[i]; a += v; q += v * v; }
    s1[tid] = a; s2[tid] = q; __syncthreads();
    for (int o = 128; o > 0; o >>= 1) {
        if (tid < o) { s1[tid] += s1[tid + o]; s2[tid] += s2[tid + o]; }
        __syncthreads();
    }
    float mu = s1[0] / H;
    float var = s2[0] / H - mu * mu;
    float rstd = rsqrtf(var + 1e-5f);
    for (int i = tid; i < H; i += 256)
        out[(long)row * H + i] = f2bf((x[i] - mu) * rstd * g[i] + b[i]);
}

// ---------------------------------------------------------------------------
// block-wide sum helper
// ---------------------------------------------------------------------------
static __device__ __forceinline__ float block_sum(float v, float* sbuf) {
    int t = threadIdx.x;
    sbuf[t] = v; __syncthreads();
    for (int o = 128; o > 0; o >>= 1) {
        if (t < o) sbuf[t] += sbuf[t + o];
        __syncthreads();
    }
    float r = sbuf[0]; __syncthreads();
    return r;
}

static __device__ __forceinline__ float hash_uniform(unsigned a, unsigned b,
                                                     unsigned c) {
    unsigned h = a * 0x9E3779B1u ^ (b + 0x7F4A7C15u) * 0x85EBCA77u ^ (c * 0xC2B2AE3Du);
    h ^= h >> 16; h *= 0x7FEB352Du; h ^= h >> 15; h *= 0x846CA68Bu; h ^= h >> 16;
    return (float)((h >> 8) | 1u) * (1.0f / 16777216.0f);  // (0,1)
}

// ---------------------------------------------------------------------------
// Residual quantize step: gumbel-categorical sample + rotation trick.
// One block (256 threads) per sample row; D == 256 (1 element/thread).
// scores[b][j] = res_b . codebook_j  (from WMMA GEMM)
// ---------------------------------------------------------------------------
__global__ __launch_bounds__(256)
void k_quantize(float* __restrict__ res, const float* __restrict__ scores,
                const float* __restrict__ cb_f32,   // [KCB][EDIM], this layer
                const float* __restrict__ cnorm,    // [KCB], this layer
                int layer, int* __restrict__ ids,
                float* __restrict__ emb_sum, float* __restrict__ embs_norm,
                float* __restrict__ quant_loss) {
    __shared__ float sred[256];
    __shared__ float svals[256];
    __shared__ int   sidx[256];
    const int b = blockIdx.x, t = threadIdx.x;

    float r = res[(long)b * EDIM + t];
    float rn = block_sum(r * r, sred);

    // gumbel-argmax over logits = -dist/TEMP
    const float* srow = scores + (long)b * KCB;
    float best = -3.4e38f; int bestj = 0;
    for (int j = t; j < KCB; j += 256) {
        float dist = (rn + cnorm[j] - 2.0f * srow[j]) * TEMP_INV;
        float u = hash_uniform((unsigned)b, (unsigned)j,
                               (unsigned)(layer * 0x01234567u + 0x9E377u));
        float v = -dist - __logf(-__logf(u));
        if (v > best) { best = v; bestj = j; }
    }
    svals[t] = best; sidx[t] = bestj; __syncthreads();
    for (int o = 128; o > 0; o >>= 1) {
        if (t < o && svals[t + o] > svals[t]) { svals[t] = svals[t + o]; sidx[t] = sidx[t + o]; }
        __syncthreads();
    }
    int id = sidx[0]; __syncthreads();
    if (t == 0) ids[b * NLAYERS + layer] = id;

    float e  = cb_f32[(long)id * EDIM + t];
    float en = block_sum(e * e, sred);

    float u_t = r * (1.0f / (sqrtf(rn) + 1e-8f));
    float q_t = e * (1.0f / (sqrtf(en) + 1e-8f));
    float w_t = u_t + q_t;
    float wn  = block_sum(w_t * w_t, sred);
    w_t *= 1.0f / fmaxf(sqrtf(wn), 1e-6f);

    float rw = block_sum(r * w_t, sred);
    float ru = block_sum(r * u_t, sred);
    float emb_out = r - 2.0f * rw * w_t + 2.0f * ru * q_t;

    float d  = r - e;
    float ls = block_sum(d * d, sred);
    if (t == 0) atomicAdd(quant_loss, (1.0f + COMMIT_W) * ls);

    res[(long)b * EDIM + t] = r - emb_out;
    emb_sum[(long)b * EDIM + t] += emb_out;

    float eo = block_sum(emb_out * emb_out, sred);
    if (t == 0) embs_norm[b * NLAYERS + layer] = sqrtf(eo);
}

// ---------------------------------------------------------------------------
// l2-normalize decoder output + recon loss; one block per row (D == 2048)
// ---------------------------------------------------------------------------
__global__ __launch_bounds__(256)
void k_recon(const float* __restrict__ xhat, const float* __restrict__ x,
             float* __restrict__ recon_sum) {
    __shared__ float sred[256];
    const int b = blockIdx.x, t = threadIdx.x;
    const float* xr = xhat + (long)b * IN_DIM;
    const float* x0 = x + (long)b * IN_DIM;
    float n2 = 0.f;
    for (int i = t; i < IN_DIM; i += 256) { float v = xr[i]; n2 += v * v; }
    n2 = block_sum(n2, sred);
    float inv = 1.0f / fmaxf(sqrtf(n2), 1e-12f);
    float acc = 0.f;
    for (int i = t; i < IN_DIM; i += 256) {
        float d = xr[i] * inv - x0[i];
        acc += d * d;
    }
    acc = block_sum(acc, sred);
    if (t == 0) atomicAdd(recon_sum, acc);
}

// ---------------------------------------------------------------------------
// p_unique: row i unique iff no j>i with all-layer id match
// ---------------------------------------------------------------------------
__global__ __launch_bounds__(256)
void k_punique(const int* __restrict__ ids, int* __restrict__ uniq) {
    __shared__ int sdup[256];
    const int i = blockIdx.x, t = threadIdx.x;
    int a0 = ids[i * 3], a1 = ids[i * 3 + 1], a2 = ids[i * 3 + 2];
    int dup = 0;
    for (int j = i + 1 + t; j < BATCH; j += 256)
        if (ids[j * 3] == a0 && ids[j * 3 + 1] == a1 && ids[j * 3 + 2] == a2) dup = 1;
    sdup[t] = dup; __syncthreads();
    for (int o = 128; o > 0; o >>= 1) {
        if (t < o) sdup[t] |= sdup[t + o];
        __syncthreads();
    }
    if (t == 0 && sdup[0] == 0) atomicAdd(uniq, 1);
}

__global__ void k_finalize(const float* __restrict__ quant_loss,
                           const float* __restrict__ recon_sum,
                           const int* __restrict__ uniq,
                           float* __restrict__ out) {
    float ql = *quant_loss;
    float rm = *recon_sum / (float)BATCH;
    out[0] = rm + ql;                 // loss = (recon + quant_loss).mean()
    out[1] = rm;                      // recon.mean()
    out[2] = ql;                      // quant_loss
    out[3 + BATCH * NLAYERS] = (float)(*uniq) / (float)BATCH;  // p_unique
}

// ---------------------------------------------------------------------------
// Host orchestration
// ---------------------------------------------------------------------------
extern "C" void kernel_launch(void* const* d_in, const int* in_sizes, int n_in,
                              void* d_out, int out_size, void* d_ws, size_t ws_size,
                              hipStream_t stream) {
    const float* x       = (const float*)d_in[0];
    const float* enc_w1  = (const float*)d_in[1];
    const float* enc_g1  = (const float*)d_in[2];
    const float* enc_b1  = (const float*)d_in[3];
    const float* enc_w2  = (const float*)d_in[4];
    const float* enc_g2  = (const float*)d_in[5];
    const float* enc_b2  = (const float*)d_in[6];
    const float* enc_w3  = (const float*)d_in[7];
    const float* dec_w1  = (const float*)d_in[8];
    const float* dec_g1  = (const float*)d_in[9];
    const float* dec_b1  = (const float*)d_in[10];
    const float* dec_w2  = (const float*)d_in[11];
    const float* dec_g2  = (const float*)d_in[12];
    const float* dec_b2  = (const float*)d_in[13];
    const float* dec_w3  = (const float*)d_in[14];
    const float* cbooks  = (const float*)d_in[15];
    float* out = (float*)d_out;

    char* ws = (char*)d_ws;
    size_t off = 0;
    auto alloc = [&](size_t bytes) -> void* {
        off = (off + 255) & ~(size_t)255;
        void* p = ws + off;
        off += bytes;
        return p;
    };

    bf16*  xbf    = (bf16*) alloc((size_t)BATCH * IN_DIM * 2);
    bf16*  w1t    = (bf16*) alloc((size_t)H1D * IN_DIM * 2);
    bf16*  w2t    = (bf16*) alloc((size_t)H2D * H1D * 2);
    bf16*  w3t    = (bf16*) alloc((size_t)EDIM * H2D * 2);
    bf16*  dw1t   = (bf16*) alloc((size_t)H2D * EDIM * 2);
    bf16*  dw2t   = (bf16*) alloc((size_t)H1D * H2D * 2);
    bf16*  dw3t   = (bf16*) alloc((size_t)IN_DIM * H1D * 2);
    bf16*  cb_bf  = (bf16*) alloc((size_t)NLAYERS * KCB * EDIM * 2);
    float* cnorm  = (float*)alloc((size_t)NLAYERS * KCB * 4);
    float* h1f    = (float*)alloc((size_t)BATCH * H1D * 4);
    bf16*  h1bf   = (bf16*) alloc((size_t)BATCH * H1D * 2);
    float* h2f    = (float*)alloc((size_t)BATCH * H2D * 4);
    bf16*  h2bf   = (bf16*) alloc((size_t)BATCH * H2D * 2);
    float* resf   = (float*)alloc((size_t)BATCH * EDIM * 4);
    bf16*  resbf  = (bf16*) alloc((size_t)BATCH * EDIM * 2);
    float* scores = (float*)alloc((size_t)BATCH * KCB * 4);
    float* embsum = (float*)alloc((size_t)BATCH * EDIM * 4);
    bf16*  esumbf = (bf16*) alloc((size_t)BATCH * EDIM * 2);
    float* d1f    = (float*)alloc((size_t)BATCH * H2D * 4);
    bf16*  d1bf   = (bf16*) alloc((size_t)BATCH * H2D * 2);
    float* d2f    = (float*)alloc((size_t)BATCH * H1D * 4);
    bf16*  d2bf   = (bf16*) alloc((size_t)BATCH * H1D * 2);
    float* d3f    = (float*)alloc((size_t)BATCH * IN_DIM * 4);
    int*   ids    = (int*)  alloc((size_t)BATCH * NLAYERS * 4);
    float* sQL    = (float*)alloc(4);
    float* sRS    = (float*)alloc(4);
    int*   sUQ    = (int*)  alloc(4);
    (void)ws_size; (void)in_sizes; (void)n_in; (void)out_size;

    // zero accumulators each launch (graph-capturable, stream-ordered)
    hipMemsetAsync(sQL, 0, 4, stream);
    hipMemsetAsync(sRS, 0, 4, stream);
    hipMemsetAsync(sUQ, 0, 4, stream);
    hipMemsetAsync(embsum, 0, (size_t)BATCH * EDIM * 4, stream);

    // convert x / weights / codebooks to bf16 (weights transposed to [N][K])
    k_f2bf<<<2048, 256, 0, stream>>>(x, xbf, (long)BATCH * IN_DIM);
    k_transpose_bf<<<2048, 256, 0, stream>>>(enc_w1, w1t, IN_DIM, H1D);
    k_transpose_bf<<<1024, 256, 0, stream>>>(enc_w2, w2t, H1D, H2D);
    k_transpose_bf<<<512,  256, 0, stream>>>(enc_w3, w3t, H2D, EDIM);
    k_transpose_bf<<<512,  256, 0, stream>>>(dec_w1, dw1t, EDIM, H2D);
    k_transpose_bf<<<1024, 256, 0, stream>>>(dec_w2, dw2t, H2D, H1D);
    k_transpose_bf<<<2048, 256, 0, stream>>>(dec_w3, dw3t, H1D, IN_DIM);
    k_f2bf<<<1024, 256, 0, stream>>>(cbooks, cb_bf, (long)NLAYERS * KCB * EDIM);
    k_rownorm2<<<NLAYERS * KCB, 256, 0, stream>>>(cbooks, cnorm, EDIM);

    // ---------------- encoder ----------------
    k_gemm_bf16_wmma<<<dim3(H1D / BN, BATCH / BM), 256, 0, stream>>>(
        xbf, w1t, h1f, BATCH, H1D, IN_DIM, 1);
    k_layernorm_bf<<<BATCH, 256, 0, stream>>>(h1f, enc_g1, enc_b1, h1bf, H1D);
    k_gemm_bf16_wmma<<<dim3(H2D / BN, BATCH / BM), 256, 0, stream>>>(
        h1bf, w2t, h2f, BATCH, H2D, H1D, 1);
    k_layernorm_bf<<<BATCH, 256, 0, stream>>>(h2f, enc_g2, enc_b2, h2bf, H2D);
    k_gemm_bf16_wmma<<<dim3(EDIM / BN, BATCH / BM), 256, 0, stream>>>(
        h2bf, w3t, resf, BATCH, EDIM, H2D, 0);

    // ---------------- residual quantization ----------------
    for (int l = 0; l < NLAYERS; ++l) {
        k_f2bf<<<1024, 256, 0, stream>>>(resf, resbf, (long)BATCH * EDIM);
        // res @ codebook.T : Bt[N][K] == codebook[l] itself
        k_gemm_bf16_wmma<<<dim3(KCB / BN, BATCH / BM), 256, 0, stream>>>(
            resbf, cb_bf + (size_t)l * KCB * EDIM, scores, BATCH, KCB, EDIM, 0);
        k_quantize<<<BATCH, 256, 0, stream>>>(
            resf, scores, cbooks + (size_t)l * KCB * EDIM,
            cnorm + (size_t)l * KCB, l, ids, embsum, out + 3, sQL);
    }

    // ---------------- decoder ----------------
    k_f2bf<<<1024, 256, 0, stream>>>(embsum, esumbf, (long)BATCH * EDIM);
    k_gemm_bf16_wmma<<<dim3(H2D / BN, BATCH / BM), 256, 0, stream>>>(
        esumbf, dw1t, d1f, BATCH, H2D, EDIM, 1);
    k_layernorm_bf<<<BATCH, 256, 0, stream>>>(d1f, dec_g1, dec_b1, d1bf, H2D);
    k_gemm_bf16_wmma<<<dim3(H1D / BN, BATCH / BM), 256, 0, stream>>>(
        d1bf, dw2t, d2f, BATCH, H1D, H2D, 1);
    k_layernorm_bf<<<BATCH, 256, 0, stream>>>(d2f, dec_g2, dec_b2, d2bf, H1D);
    k_gemm_bf16_wmma<<<dim3(IN_DIM / BN, BATCH / BM), 256, 0, stream>>>(
        d2bf, dw3t, d3f, BATCH, IN_DIM, H1D, 0);

    // ---------------- losses / stats ----------------
    k_recon<<<BATCH, 256, 0, stream>>>(d3f, x, sRS);
    k_punique<<<BATCH, 256, 0, stream>>>(ids, sUQ);
    k_finalize<<<1, 1, 0, stream>>>(sQL, sRS, sUQ, out);
}
